// ScaledDotProductAttention_6493990552302
// MI455X (gfx1250) — matrix operand order repaired and checked by reading, not compiled, not run on hardware
//
#include <hip/hip_runtime.h>

typedef __attribute__((ext_vector_type(16))) __bf16 bf16x16;
typedef __attribute__((ext_vector_type(8)))  __bf16 bf16x8;
typedef __attribute__((ext_vector_type(8)))  float  f32x8;
typedef __attribute__((ext_vector_type(4)))  float  f32x4;

namespace {
constexpr int   kB      = 4;
constexpr int   kH      = 16;
constexpr int   kS      = 2048;
constexpr int   kD      = 64;
constexpr int   kBN     = 64;   // keys per iteration
constexpr int   kStride = 72;   // LDS row stride in bf16 elements (64 + 8 pad)
constexpr float kScale  = 0.08838834764831845f;  // 1/sqrt(64*2)
}

__device__ __forceinline__ f32x8 wmma_bf16(bf16x16 a, bf16x16 b, f32x8 c) {
  return __builtin_amdgcn_wmma_f32_16x16x32_bf16(false, a, false, b, (short)0, c,
                                                 false, false);
}

// A-fragment (16x32, 16-bit): lanes 0-15 hold K {k0+0..7, k0+16..23},
// lanes 16-31 hold K {k0+8..15, k0+24..31} of row M = lane&15.
__device__ __forceinline__ bf16x16 frag_a_lds(const __bf16* row, int k0, int hl) {
  bf16x8 lo = *(const bf16x8*)(row + k0 + hl * 8);
  bf16x8 hi = *(const bf16x8*)(row + k0 + 16 + hl * 8);
  bf16x16 r;
#pragma unroll
  for (int i = 0; i < 8; ++i) { r[i] = lo[i]; r[8 + i] = hi[i]; }
  return r;
}

// B-fragment (32x16, 16-bit): lane holds column N = lane&15; lanes 0-15 hold
// K {k0+0..7, k0+16..23}, lanes 16-31 hold K {k0+8..15, k0+24..31}.
__device__ __forceinline__ bf16x16 frag_b_lds(const __bf16* row, int k0, int hl) {
  bf16x8 lo = *(const bf16x8*)(row + k0 + hl * 8);
  bf16x8 hi = *(const bf16x8*)(row + k0 + 16 + hl * 8);
  bf16x16 r;
#pragma unroll
  for (int i = 0; i < 8; ++i) { r[i] = lo[i]; r[8 + i] = hi[i]; }
  return r;
}

// A-fragment built directly from a global fp32 row (for Q, loaded once).
__device__ __forceinline__ bf16x16 frag_a_global(const float* row, int k0, int hl) {
  f32x4 a0 = *(const f32x4*)(row + k0 + hl * 8);
  f32x4 a1 = *(const f32x4*)(row + k0 + hl * 8 + 4);
  f32x4 b0 = *(const f32x4*)(row + k0 + 16 + hl * 8);
  f32x4 b1 = *(const f32x4*)(row + k0 + 16 + hl * 8 + 4);
  bf16x16 r;
#pragma unroll
  for (int i = 0; i < 4; ++i) {
    r[i]      = (__bf16)a0[i];
    r[4 + i]  = (__bf16)a1[i];
    r[8 + i]  = (__bf16)b0[i];
    r[12 + i] = (__bf16)b1[i];
  }
  return r;
}

__global__ __launch_bounds__(128) void flash_attn_bf16_wmma(
    const float* __restrict__ Q, const float* __restrict__ K,
    const float* __restrict__ V, const int* __restrict__ mask,
    float* __restrict__ out) {
  __shared__ __bf16 Ks[kBN][kStride];        // key-major, row = key, col = d
  __shared__ __bf16 Vt[kD][kStride];         // transposed: row = d, col = key
  __shared__ __bf16 Ps[4][16][kStride];      // per-wave P staging, row = m, col = key

  const int tid  = threadIdx.x;
  const int wave = tid >> 5;    // 0..3
  const int lane = tid & 31;
  const int hl   = lane >> 4;   // lane half (0: lanes 0-15, 1: lanes 16-31)
  const int lm   = lane & 15;

  const int mblk = blockIdx.x & 31;   // S/64 = 32 query blocks
  const int bh   = blockIdx.x >> 5;   // 0..63 (b*H + h)
  const int b    = bh >> 4;           // H = 16

  const size_t base = (size_t)bh * kS * kD;
  const float* Qb = Q + base;
  const float* Kb = K + base;
  const float* Vb = V + base;
  const int*   Mb = mask + (size_t)b * kS * kS;
  float*       Ob = out + base;

  const int m0 = mblk * 64 + wave * 16;   // this wave's 16 query rows

  // Q fragments (persist for whole kernel)
  const float* qrow = Qb + (size_t)(m0 + lm) * kD;
  const bf16x16 qa0 = frag_a_global(qrow, 0, hl);
  const bf16x16 qa1 = frag_a_global(qrow, 32, hl);

  f32x8 acc[4] = {{0,0,0,0,0,0,0,0},{0,0,0,0,0,0,0,0},
                  {0,0,0,0,0,0,0,0},{0,0,0,0,0,0,0,0}};
  float mi[8], li[8];
#pragma unroll
  for (int r = 0; r < 8; ++r) { mi[r] = -__builtin_inff(); li[r] = 0.0f; }

  // cooperative-load assignment: 2 threads per key row, 32 d-elems each.
  // 128 threads x 32 elems = 4096 = full 64x64 tile.
  const int krow = tid >> 1;          // 0..63
  const int kd0  = (tid & 1) * 32;    // 0 or 32

  for (int n0 = 0; n0 < kS; n0 += kBN) {
    __syncthreads();  // previous iteration's LDS reads complete

    // ---- fill K tile (bf16, row-major) and V tile (bf16, transposed) ----
    {
      const float* ks = Kb + (size_t)(n0 + krow) * kD + kd0;
      f32x4 kv[8];
#pragma unroll
      for (int j = 0; j < 8; ++j) kv[j] = ((const f32x4*)ks)[j];
#pragma unroll
      for (int j = 0; j < 4; ++j) {
        bf16x8 p;
#pragma unroll
        for (int i = 0; i < 4; ++i) {
          p[i]     = (__bf16)kv[2 * j][i];
          p[4 + i] = (__bf16)kv[2 * j + 1][i];
        }
        *(bf16x8*)&Ks[krow][kd0 + 8 * j] = p;
      }

      const float* vs = Vb + (size_t)(n0 + krow) * kD + kd0;
      f32x4 vv[8];
#pragma unroll
      for (int j = 0; j < 8; ++j) vv[j] = ((const f32x4*)vs)[j];
#pragma unroll
      for (int j = 0; j < 8; ++j) {
#pragma unroll
        for (int i = 0; i < 4; ++i) {
          Vt[kd0 + 4 * j + i][krow] = (__bf16)vv[j][i];
        }
      }

      if (n0 + kBN < kS) {  // prefetch next tiles into cache hierarchy
        __builtin_prefetch(Kb + (size_t)(n0 + kBN + krow) * kD + kd0, 0, 0);
        __builtin_prefetch(Vb + (size_t)(n0 + kBN + krow) * kD + kd0, 0, 0);
      }
    }
    __syncthreads();

    // ---- scores: S = Q * K^T  (4 n-tiles, K-dim 64 = 2 chained WMMAs) ----
    f32x8 st[4];
#pragma unroll
    for (int c = 0; c < 4; ++c) {
      const __bf16* kr = &Ks[c * 16 + lm][0];
      bf16x16 kb0 = frag_b_lds(kr, 0, hl);
      bf16x16 kb1 = frag_b_lds(kr, 32, hl);
      f32x8 z = {0, 0, 0, 0, 0, 0, 0, 0};
      z = wmma_bf16(qa0, kb0, z);
      z = wmma_bf16(qa1, kb1, z);
      st[c] = z;
    }

    // ---- scale + mask (reference: scale first, then mask==0 -> -1e9) ----
#pragma unroll
    for (int c = 0; c < 4; ++c) {
#pragma unroll
      for (int r = 0; r < 8; ++r) {
        float s = st[c][r] * kScale;
        const int q = m0 + hl * 8 + r;
        const int k = n0 + c * 16 + lm;
        if (Mb[(size_t)q * kS + k] == 0) s = -1.0e9f;
        st[c][r] = s;
      }
    }

    // ---- online softmax: row max over 64 cols (16 lanes x 4 tiles) ----
    float al[8];
#pragma unroll
    for (int r = 0; r < 8; ++r) {
      float v = fmaxf(fmaxf(st[0][r], st[1][r]), fmaxf(st[2][r], st[3][r]));
      v = fmaxf(v, __shfl_xor(v, 1, 32));
      v = fmaxf(v, __shfl_xor(v, 2, 32));
      v = fmaxf(v, __shfl_xor(v, 4, 32));
      v = fmaxf(v, __shfl_xor(v, 8, 32));
      const float mn = fmaxf(mi[r], v);
      al[r] = __expf(mi[r] - mn);
      mi[r] = mn;
    }

    // p = exp(s - m_new); stage bf16 P into LDS in A-readable row-major form
    float rsum[8] = {0, 0, 0, 0, 0, 0, 0, 0};
#pragma unroll
    for (int c = 0; c < 4; ++c) {
#pragma unroll
      for (int r = 0; r < 8; ++r) {
        const float p = __expf(st[c][r] - mi[r]);
        rsum[r] += p;
        Ps[wave][hl * 8 + r][c * 16 + lm] = (__bf16)p;
      }
    }
#pragma unroll
    for (int r = 0; r < 8; ++r) {
      float v = rsum[r];
      v += __shfl_xor(v, 1, 32);
      v += __shfl_xor(v, 2, 32);
      v += __shfl_xor(v, 4, 32);
      v += __shfl_xor(v, 8, 32);
      li[r] = li[r] * al[r] + v;
    }
#pragma unroll
    for (int dc = 0; dc < 4; ++dc) {
#pragma unroll
      for (int r = 0; r < 8; ++r) acc[dc][r] *= al[r];
    }

    // wave-internal LDS sync: P stores visible to cross-lane A-fragment loads
    asm volatile("s_wait_dscnt 0x0" ::: "memory");

    // ---- context += P * V  (4 d-tiles, K-dim 64 = 2 chained WMMAs) ----
    const bf16x16 pa0 = frag_a_lds(&Ps[wave][lm][0], 0, hl);
    const bf16x16 pa1 = frag_a_lds(&Ps[wave][lm][0], 32, hl);
#pragma unroll
    for (int dc = 0; dc < 4; ++dc) {
      const __bf16* vr = &Vt[dc * 16 + lm][0];
      bf16x16 vb0 = frag_b_lds(vr, 0, hl);
      bf16x16 vb1 = frag_b_lds(vr, 32, hl);
      acc[dc] = wmma_bf16(pa0, vb0, acc[dc]);
      acc[dc] = wmma_bf16(pa1, vb1, acc[dc]);
    }
  }

  // ---- epilogue: normalize by row sum, store fp32 context ----
#pragma unroll
  for (int r = 0; r < 8; ++r) {
    const float inv = 1.0f / li[r];
    const int q = m0 + hl * 8 + r;
#pragma unroll
    for (int dc = 0; dc < 4; ++dc) {
      Ob[(size_t)q * kD + dc * 16 + lm] = acc[dc][r] * inv;
    }
  }
}

extern "C" void kernel_launch(void* const* d_in, const int* in_sizes, int n_in,
                              void* d_out, int out_size, void* d_ws, size_t ws_size,
                              hipStream_t stream) {
  (void)in_sizes; (void)n_in; (void)out_size; (void)d_ws; (void)ws_size;
  const float* Q    = (const float*)d_in[0];
  const float* K    = (const float*)d_in[1];
  const float* V    = (const float*)d_in[2];
  // d_in[3] (distance_matrix) and d_in[4] (energy_matrix) are unused by the
  // reference forward.
  const int*   mask = (const int*)d_in[5];
  float*       out  = (float*)d_out;

  dim3 grid(kB * kH * (kS / 64));   // 2048 workgroups, one per 64-row query block
  dim3 block(128);                  // 4 wave32 waves; wave w owns query rows
                                    // [mblk*64 + w*16, +16)
  flash_attn_bf16_wmma<<<grid, block, 0, stream>>>(Q, K, V, mask, out);
}
